// MPNNLayer_38878043963479
// MI455X (gfx1250) — matrix operand order repaired and checked by reading, not compiled, not run on hardware
//
#include <hip/hip_runtime.h>
#include <hip/hip_bf16.h>

typedef __attribute__((ext_vector_type(16))) _Float16 v16h;
typedef __attribute__((ext_vector_type(8)))  _Float16 v8h;
typedef __attribute__((ext_vector_type(8)))  float    v8f;

#define N_NODES  50000
#define N_EDGES  800000
#define EDGE_DIM 16
#define MLP_HID  64
#define MSG_DIM  16
#define HID_DIM  16

#define WAVES 8
#define XP   72    // x row pitch (halves): 144B = 36-dword lane stride -> conflict-free
#define W2P  72    // W2^T row pitch (halves)

// single-instruction relu: v_med3_f32(x, 0, FLT_MAX) -- avoids the max(x,x)+max(0,x)
// canonicalization pair that fmaxf/ternary generate
__device__ __forceinline__ float relu1(float x) {
  return __builtin_amdgcn_fmed3f(x, 0.0f, 3.402823466e38f);
}

// 16 consecutive halves (two ds_load_b128) -> fragment
__device__ __forceinline__ v16h ld_frag(const _Float16* p) {
  v8h lo = *(const v8h*)p;
  v8h hi = *(const v8h*)(p + 8);
  v16h r;
#pragma unroll
  for (int i = 0; i < 8; ++i) { r[i] = lo[i]; r[8 + i] = hi[i]; }
  return r;
}

// two disjoint 8-half groups (A-fragment element halves) -> fragment
__device__ __forceinline__ v16h ld_frag2(const _Float16* p0, const _Float16* p1) {
  v8h lo = *(const v8h*)p0;
  v8h hi = *(const v8h*)p1;
  v16h r;
#pragma unroll
  for (int i = 0; i < 8; ++i) { r[i] = lo[i]; r[8 + i] = hi[i]; }
  return r;
}

__global__ void zero_kernel(float* __restrict__ p, int n) {
  int i = blockIdx.x * blockDim.x + threadIdx.x;
  if (i < n) p[i] = 0.0f;
}

// Transposed formulation: one wave owns 16 edges (edge = N dimension everywhere).
//   x^T = relu(W1^T @ e^T + b1)   64x16  (bias rides in the WMMA C operand)
//   A^T = relu(W2^T @ x^T + b2)  256x16
//   msg[e][nt] = sum_h A[e][nt*16+h] * h[src[e]][h]   (in-register FMA + 1 shfl_xor)
//   atomicAdd into m[dst[e]]
// __launch_bounds__(256, 4): >=4 waves/SIMD => <=256 VGPRs, no MSB addressing/spills.
__global__ __launch_bounds__(WAVES * 32, 4) void edge_kernel(
    const float* __restrict__ e,  const float* __restrict__ h,
    const float* __restrict__ W1, const float* __restrict__ b1,
    const float* __restrict__ W2, const float* __restrict__ b2,
    const int* __restrict__ src,  const int* __restrict__ dst,
    float* __restrict__ m_out, int n_tiles)
{
  __shared__ _Float16 w1h[64 * 32];            // W1^T rows n, K 0..31 (16..31 zero) 4KB
  __shared__ _Float16 w2h[256 * W2P];          // W2^T rows n, K 0..63             36KB
  __shared__ float    b1s[64];
  __shared__ float    b2s[256];
  __shared__ _Float16 xbuf[WAVES][16 * XP];    // per-wave x rows = edges          18KB

  const int tid = threadIdx.x;

  // ---- stage W1^T/W2^T (f16) + biases into LDS
  for (int i = tid; i < 64 * 32; i += blockDim.x) {
    int n = i >> 5, k = i & 31;
    w1h[i] = (k < EDGE_DIM) ? (_Float16)W1[k * MLP_HID + n] : (_Float16)0.0f;
  }
  for (int i = tid; i < 256 * 64; i += blockDim.x) {
    int n = i >> 6, k = i & 63;
    w2h[n * W2P + k] = (_Float16)W2[k * 256 + n];
  }
  for (int i = tid; i < 64; i += blockDim.x)  b1s[i] = b1[i];
  for (int i = tid; i < 256; i += blockDim.x) b2s[i] = b2[i];
  __syncthreads();

  const int lane   = tid & 31;
  const int wid    = tid >> 5;
  const int halfid = lane >> 4;   // 0: lanes 0-15, 1: lanes 16-31
  const int myh    = lane & 15;   // this lane's edge within the tile (C/B column)
  _Float16* xb = xbuf[wid];

  // ---- preload W1^T A-fragments (rows 16t+myh; elem 0..15 = K 16h..16h+15)
  v16h aw1[4];
#pragma unroll
  for (int t = 0; t < 4; ++t) {
    const _Float16* p = &w1h[(16 * t + myh) * 32 + 16 * halfid];
    aw1[t] = ld_frag2(p, p + 8);
  }
  // ---- preload b1 as WMMA C-init (C rows n = 16t + v + 8*halfid)
  v8f b1c[4];
#pragma unroll
  for (int t = 0; t < 4; ++t)
    b1c[t] = *(const v8f*)&b1s[16 * t + 8 * halfid];

  const int gwave  = blockIdx.x * WAVES + wid;
  const int stride = gridDim.x * WAVES;

  for (int tile = gwave; tile < n_tiles; tile += stride) {
    const int ebase = tile * 16;
    const int sval = src[ebase + myh];   // own edge's endpoints (no shuffles needed)
    const int dval = dst[ebase + myh];

    // ---- B fragment of e^T: lane col = edge myh; lanes<16 carry K 0..15, lanes>=16 = pad
    v16h be;
    if (halfid == 0) {
      const float* ep = e + (size_t)(ebase + myh) * EDGE_DIM;
      float4 f0 = *(const float4*)ep;
      float4 f1 = *(const float4*)(ep + 4);
      float4 f2 = *(const float4*)(ep + 8);
      float4 f3 = *(const float4*)(ep + 12);
      be[0]  = (_Float16)f0.x; be[1]  = (_Float16)f0.y;
      be[2]  = (_Float16)f0.z; be[3]  = (_Float16)f0.w;
      be[4]  = (_Float16)f1.x; be[5]  = (_Float16)f1.y;
      be[6]  = (_Float16)f1.z; be[7]  = (_Float16)f1.w;
      be[8]  = (_Float16)f2.x; be[9]  = (_Float16)f2.y;
      be[10] = (_Float16)f2.z; be[11] = (_Float16)f2.w;
      be[12] = (_Float16)f3.x; be[13] = (_Float16)f3.y;
      be[14] = (_Float16)f3.z; be[15] = (_Float16)f3.w;
      // stream hint for the next tile this wave will touch
      __builtin_prefetch(e + (size_t)(ebase + stride * 16 + myh) * EDGE_DIM, 0, 0);
    } else {
#pragma unroll
      for (int i = 0; i < 16; ++i) be[i] = (_Float16)0.0f;
    }

    // ---- GEMM1': x^T = relu(W1^T @ e^T + b1); lane's 8 outputs are contiguous n for
    //      its own edge -> one packed ds_store_b128 per tile
#pragma unroll
    for (int t = 0; t < 4; ++t) {
      v8f c = __builtin_amdgcn_wmma_f32_16x16x32_f16(false, aw1[t], false, be,
                                                     (short)0, b1c[t], false, false);
      v8h hx;
#pragma unroll
      for (int v = 0; v < 8; ++v)
        hx[v] = (_Float16)relu1(c[v]);           // v_med3 + v_cvt_pk pairs
      *(v8h*)&xb[myh * XP + 16 * t + 8 * halfid] = hx;
    }

    // ---- B fragments of x^T (row-major x, contiguous K; same-wave DS is in-order)
    v16h bx[2];
#pragma unroll
    for (int cc = 0; cc < 2; ++cc)
      bx[cc] = ld_frag(&xb[myh * XP + 32 * cc + 16 * halfid]);

    // ---- h[src] slice for this lane: hdim = 8*halfid + v (contiguous -> 2x float4)
    const float* hp = h + (size_t)sval * HID_DIM + 8 * halfid;
    float4 h0 = *(const float4*)hp;
    float4 h1 = *(const float4*)(hp + 4);
    float hsr[8] = {h0.x, h0.y, h0.z, h0.w, h1.x, h1.y, h1.z, h1.w};

    // ---- GEMM2' over even/odd nt pairs: relu * h_src, 16-wide reduce, immediate
    //      scatter-add (half0 owns even nt, half1 odd). No indexed private arrays.
#pragma unroll 2
    for (int i = 0; i < 8; ++i) {
      float acc0, acc1;
#pragma unroll
      for (int p = 0; p < 2; ++p) {
        const int nt = 2 * i + p;
        v8f c = *(const v8f*)&b2s[16 * nt + 8 * halfid];   // bias as C-init
#pragma unroll
        for (int cc = 0; cc < 2; ++cc) {
          const _Float16* q = &w2h[(16 * nt + myh) * W2P + 32 * cc + 16 * halfid];
          v16h a = ld_frag2(q, q + 8);
          c = __builtin_amdgcn_wmma_f32_16x16x32_f16(false, a, false, bx[cc],
                                                     (short)0, c, false, false);
        }
        float acc = 0.0f;
#pragma unroll
        for (int v = 0; v < 8; ++v)
          acc = fmaf(relu1(c[v]), hsr[v], acc);   // relu * h[src][hdim], partial sum
        acc += __shfl_xor(acc, 16, 32);           // combine hdim halves
        if (p == 0) acc0 = acc; else acc1 = acc;
      }
      const float val = halfid ? acc1 : acc0;
      atomicAdd(&m_out[(size_t)dval * MSG_DIM + 2 * i + halfid], val);
    }
  }
}

// one thread per node: gi = m@W_ih^T + b_ih ; gh = h@W_hh^T + b_hh ; GRU gates (r,z,n)
__global__ __launch_bounds__(128) void gru_kernel(
    const float* __restrict__ m,    const float* __restrict__ h,
    const float* __restrict__ W_ih, const float* __restrict__ W_hh,
    const float* __restrict__ b_ih, const float* __restrict__ b_hh,
    float* __restrict__ out, int n_nodes)
{
  __shared__ float wih[48 * 16], whh[48 * 16], bih[48], bhh[48];
  for (int i = threadIdx.x; i < 48 * 16; i += blockDim.x) { wih[i] = W_ih[i]; whh[i] = W_hh[i]; }
  for (int i = threadIdx.x; i < 48; i += blockDim.x)      { bih[i] = b_ih[i]; bhh[i] = b_hh[i]; }
  __syncthreads();

  const int node = blockIdx.x * blockDim.x + threadIdx.x;
  if (node >= n_nodes) return;

  float mv[16], hv[16];
#pragma unroll
  for (int k = 0; k < 16; ++k) {
    mv[k] = m[(size_t)node * 16 + k];
    hv[k] = h[(size_t)node * 16 + k];
  }
#pragma unroll 4
  for (int j = 0; j < 16; ++j) {
    float ir = bih[j], iz = bih[16 + j], in_ = bih[32 + j];
    float hr = bhh[j], hz = bhh[16 + j], hn  = bhh[32 + j];
#pragma unroll
    for (int k = 0; k < 16; ++k) {
      ir  += wih[j * 16 + k] * mv[k];
      iz  += wih[(16 + j) * 16 + k] * mv[k];
      in_ += wih[(32 + j) * 16 + k] * mv[k];
      hr  += whh[j * 16 + k] * hv[k];
      hz  += whh[(16 + j) * 16 + k] * hv[k];
      hn  += whh[(32 + j) * 16 + k] * hv[k];
    }
    const float r = 1.0f / (1.0f + __expf(-(ir + hr)));
    const float z = 1.0f / (1.0f + __expf(-(iz + hz)));
    const float n = tanhf(in_ + r * hn);
    out[(size_t)node * 16 + j] = (1.0f - z) * n + z * hv[j];
  }
}

extern "C" void kernel_launch(void* const* d_in, const int* in_sizes, int n_in,
                              void* d_out, int out_size, void* d_ws, size_t ws_size,
                              hipStream_t stream) {
  (void)in_sizes; (void)n_in; (void)out_size; (void)ws_size;
  const float* e   = (const float*)d_in[0];
  const float* h   = (const float*)d_in[1];
  const float* W1  = (const float*)d_in[2];
  const float* b1  = (const float*)d_in[3];
  const float* W2  = (const float*)d_in[4];
  const float* b2  = (const float*)d_in[5];
  const float* Wih = (const float*)d_in[6];
  const float* Whh = (const float*)d_in[7];
  const float* bih = (const float*)d_in[8];
  const float* bhh = (const float*)d_in[9];
  const int*   src = (const int*)d_in[10];
  const int*   dst = (const int*)d_in[11];
  float* out  = (float*)d_out;
  float* m_ws = (float*)d_ws;   // [N_NODES, MSG_DIM] accumulator

  const int mcount = N_NODES * MSG_DIM;
  zero_kernel<<<(mcount + 255) / 256, 256, 0, stream>>>(m_ws, mcount);

  const int n_tiles = N_EDGES / 16;  // 50,000 (E is an exact multiple of 16)
  edge_kernel<<<640, WAVES * 32, 0, stream>>>(
      e, h, W1, b1, W2, b2, src, dst, m_ws, n_tiles);

  gru_kernel<<<(N_NODES + 127) / 128, 128, 0, stream>>>(
      m_ws, h, Wih, Whh, bih, bhh, out, N_NODES);
}
